// Seq2Seq_19567871001157
// MI455X (gfx1250) — compile-verified, matrix operand order checked by reading, not run on hardware
//
#include <hip/hip_runtime.h>
#include <hip/hip_bf16.h>
#include <math.h>

// ---------------------------------------------------------------------------
// Types for CDNA5 WMMA (wave32): D(16x16,f32) = A(16x32,f16) * B(32x16,f16) + C
// ---------------------------------------------------------------------------
typedef __attribute__((ext_vector_type(16))) _Float16 v16h;
typedef __attribute__((ext_vector_type(8)))  _Float16 v8h;
typedef __attribute__((ext_vector_type(8)))  float    v8f;

static constexpr int S     = 256;
static constexpr int B     = 128;
static constexpr int H     = 768;
static constexpr int STUFF = 22;
static constexpr int NUM   = 15;
static constexpr int V     = STUFF + NUM;   // 37
static constexpr int T1    = 45;            // T - 1 decode steps
static constexpr int H2    = 2 * H;         // 1536
static constexpr int H3    = 3 * H;         // 2304
static constexpr float NEGV = -1.0e12f;

__device__ __forceinline__ float sigmoidf_(float x) { return 1.0f / (1.0f + expf(-x)); }

__device__ __forceinline__ v16h cat16h(v8h lo, v8h hi) {
    return __builtin_shufflevector(lo, hi, 0,1,2,3,4,5,6,7,8,9,10,11,12,13,14,15);
}

// ---------------------------------------------------------------------------
// f16 WMMA GEMM:  C[M,N] = act( A16[M,K] * Wt16[N,K]^T + bias[N] )
//   A16 row-major [M][K] f16, Wt16 row-major [N][K] f16 (W transposed).
//   Each wave computes a 16x32 tile (one A fragment shared by two B tiles ->
//   two independent WMMA accumulator chains). Block = 4 waves = 32x64 tile.
//   Loads for chunk k+1 are issued before the WMMAs of chunk k (software
//   pipeline) so v_wmma issues while the next global_load_b128s are in flight.
//   Requires: M % 32 == 0, N % 64 == 0, K % 32 == 0 (true for all GEMMs here).
//   Fragment layout (ISA 7.12.2, 16-bit A 16x32): lane l holds row m0+(l&15);
//   elems 0..7 -> k = 8g+0..7, elems 8..15 -> k = 16+8g+0..7 (g = l>>4).
//   B fragment identical with n in place of m => all loads are contiguous 16B.
// ---------------------------------------------------------------------------
struct Frag { v8h alo, ahi, b0lo, b0hi, b1lo, b1hi; };

__device__ __forceinline__ Frag load_frag(const _Float16* arow,
                                          const _Float16* brow0,
                                          const _Float16* brow1,
                                          int kc, int g) {
    Frag f;
    f.alo  = *(const v8h*)(arow  + kc + 8 * g);
    f.ahi  = *(const v8h*)(arow  + kc + 16 + 8 * g);
    f.b0lo = *(const v8h*)(brow0 + kc + 8 * g);
    f.b0hi = *(const v8h*)(brow0 + kc + 16 + 8 * g);
    f.b1lo = *(const v8h*)(brow1 + kc + 8 * g);
    f.b1hi = *(const v8h*)(brow1 + kc + 16 + 8 * g);
    return f;
}

__launch_bounds__(128)
__global__ void gemm_wmma_f16(const _Float16* __restrict__ A,
                              const _Float16* __restrict__ Wt,
                              const float* __restrict__ bias,
                              float* __restrict__ Cf,
                              _Float16* __restrict__ Ch,
                              int M, int N, int K, int act /*0=none,1=tanh*/) {
    const int wave = threadIdx.x >> 5;        // 0..3
    const int lane = threadIdx.x & 31;
    const int g    = lane >> 4;               // half-wave: selects k-subgroup
    const int ln   = lane & 15;

    const int m0 = blockIdx.y * 32 + (wave >> 1) * 16;
    const int n0 = blockIdx.x * 64 + (wave & 1) * 32;   // wave covers [n0, n0+32)

    const _Float16* arow  = A  + (size_t)(m0 + ln) * K;
    const _Float16* brow0 = Wt + (size_t)(n0 + ln) * K;
    const _Float16* brow1 = Wt + (size_t)(n0 + 16 + ln) * K;

    v8f acc0 = {}, acc1 = {};

    Frag cur = load_frag(arow, brow0, brow1, 0, g);
    for (int kc = 32; kc < K; kc += 32) {
        __builtin_prefetch(brow0 + kc + 32, 0, 1);  // speculative; dropped if OOB
        __builtin_prefetch(brow1 + kc + 32, 0, 1);
        Frag nxt = load_frag(arow, brow0, brow1, kc, g);   // loads in flight...
        v16h a  = cat16h(cur.alo,  cur.ahi);
        v16h b0 = cat16h(cur.b0lo, cur.b0hi);
        v16h b1 = cat16h(cur.b1lo, cur.b1hi);
        acc0 = __builtin_amdgcn_wmma_f32_16x16x32_f16(false, a, false, b0,
                                                      (short)0, acc0, false, false);
        acc1 = __builtin_amdgcn_wmma_f32_16x16x32_f16(false, a, false, b1,
                                                      (short)0, acc1, false, false);
        cur = nxt;                                          // ...while WMMAs execute
    }
    {
        v16h a  = cat16h(cur.alo,  cur.ahi);
        v16h b0 = cat16h(cur.b0lo, cur.b0hi);
        v16h b1 = cat16h(cur.b1lo, cur.b1hi);
        acc0 = __builtin_amdgcn_wmma_f32_16x16x32_f16(false, a, false, b0,
                                                      (short)0, acc0, false, false);
        acc1 = __builtin_amdgcn_wmma_f32_16x16x32_f16(false, a, false, b1,
                                                      (short)0, acc1, false, false);
    }

    // C layout: VGPR j -> M = m0 + j + 8g, N = n + (lane&15)
    const int nA = n0 + ln;
    const int nB = n0 + 16 + ln;
    const float bbA = bias ? bias[nA] : 0.0f;
    const float bbB = bias ? bias[nB] : 0.0f;
#pragma unroll
    for (int j = 0; j < 8; ++j) {
        const int m = m0 + j + 8 * g;
        float v0 = acc0[j] + bbA;
        float v1 = acc1[j] + bbB;
        if (act == 1) { v0 = tanhf(v0); v1 = tanhf(v1); }
        if (Cf) { Cf[(size_t)m * N + nA] = v0; Cf[(size_t)m * N + nB] = v1; }
        if (Ch) { Ch[(size_t)m * N + nA] = (_Float16)v0; Ch[(size_t)m * N + nB] = (_Float16)v1; }
    }
}

// ---------------------------------------------------------------------------
// Precompute kernels
// ---------------------------------------------------------------------------
// W[K][N] f32 -> Wt[N][K] f16
__global__ void k_wt_convert(_Float16* __restrict__ dst, const float* __restrict__ src,
                             int K, int N) {
    long i = (long)blockIdx.x * blockDim.x + threadIdx.x;
    if (i >= (long)K * N) return;
    int k = (int)(i / N), n = (int)(i % N);
    dst[(size_t)n * K + k] = (_Float16)src[i];
}

// enc16[(b*S+s)*H+h] = encoder_out[s][b][h]  (transpose + f16 convert)
__global__ void k_convert_enc(_Float16* __restrict__ enc16,
                              const float* __restrict__ enc_out) {
    long i = (long)blockIdx.x * blockDim.x + threadIdx.x;
    if (i >= (long)B * S * H) return;
    int b = (int)(i / ((long)S * H));
    int r = (int)(i % ((long)S * H));
    int s = r / H, h = r % H;
    enc16[i] = (_Float16)enc_out[((size_t)s * B + b) * H + h];
}

// wdv[b][v][h]: v<22 -> embedding row v; else gathered/zeroed num embedding.
__global__ void k_build_wdv(float* __restrict__ wdv, _Float16* __restrict__ wdv16,
                            const float* __restrict__ embw,
                            const float* __restrict__ enc_out,
                            const int* __restrict__ num_pos,
                            const int* __restrict__ num_count) {
    long i = (long)blockIdx.x * blockDim.x + threadIdx.x;
    if (i >= (long)B * V * H) return;
    int b = (int)(i / ((long)V * H));
    int r = (int)(i % ((long)V * H));
    int v = r / H, h = r % H;
    float val;
    if (v < STUFF) {
        val = embw[(size_t)v * H + h];
    } else {
        int j = v - STUFF;
        int cnt = num_count[b]; if (cnt < 1) cnt = 1;
        if (j < cnt) {
            int p = num_pos[b * NUM + j];
            val = enc_out[((size_t)p * B + b) * H + h];
        } else {
            val = 0.0f;
        }
    }
    wdv[i] = val;
    wdv16[i] = (_Float16)val;
}

__global__ void k_build_mask(unsigned char* __restrict__ mask,
                             const int* __restrict__ num_count) {
    int i = blockIdx.x * blockDim.x + threadIdx.x;
    if (i >= B * V) return;
    int b = i / V, v = i % V;
    int cnt = num_count[b]; if (cnt < 1) cnt = 1;
    mask[i] = (v >= STUFF && (v - STUFF) >= cnt) ? 1 : 0;
}

// h=0, h16=0, feed=relu(problem), inp=0
__global__ void k_init(float* __restrict__ h, _Float16* __restrict__ h16,
                       float* __restrict__ feed, int* __restrict__ inp,
                       const float* __restrict__ problem) {
    int i = blockIdx.x * blockDim.x + threadIdx.x;
    if (i >= B * H) return;
    h[i] = 0.0f;
    h16[i] = (_Float16)0.0f;
    feed[i] = fmaxf(problem[i], 0.0f);
    if (i < B) inp[i] = 0;
}

// ---------------------------------------------------------------------------
// Per-step kernels
// ---------------------------------------------------------------------------
// One block per batch b (256 threads = 8 waves, one thread per s).
// e[s] = tanh(enc_proj16[b,s,:] + hp[b,:]) . attn_v ; mask ; softmax ; ctx.
__launch_bounds__(256)
__global__ void k_attention(const _Float16* __restrict__ encproj16,
                            const float* __restrict__ hp,
                            const float* __restrict__ attn_v,
                            const int* __restrict__ input_length,
                            const float* __restrict__ enc_out,
                            float* __restrict__ ctx) {
    __shared__ float hp_s[H];
    __shared__ float av_s[H];
    __shared__ float w_s[S];
    __shared__ float red[256];
    const int b = blockIdx.x;
    const int tid = threadIdx.x;

    for (int i = tid; i < H; i += 256) { hp_s[i] = hp[(size_t)b * H + i]; av_s[i] = attn_v[i]; }
    __syncthreads();

    int len = input_length[b]; if (len < 1) len = 1;
    float e = NEGV;
    if (tid < len) {
        const v8h* row8 = (const v8h*)(encproj16 + ((size_t)(b * S + tid)) * H);
        float acc = 0.0f;
        for (int c = 0; c < H / 8; ++c) {
            v8h rv = row8[c];
#pragma unroll
            for (int j = 0; j < 8; ++j)
                acc = fmaf(tanhf((float)rv[j] + hp_s[c * 8 + j]), av_s[c * 8 + j], acc);
        }
        e = acc;
    }
    // softmax over S=256 lanes (block reduce max then sum)
    red[tid] = e; __syncthreads();
    for (int st = 128; st > 0; st >>= 1) { if (tid < st) red[tid] = fmaxf(red[tid], red[tid + st]); __syncthreads(); }
    const float mx = red[0]; __syncthreads();
    const float ex = expf(e - mx);
    w_s[tid] = ex;
    red[tid] = ex; __syncthreads();
    for (int st = 128; st > 0; st >>= 1) { if (tid < st) red[tid] += red[tid + st]; __syncthreads(); }
    const float inv = 1.0f / red[0]; __syncthreads();

    // ctx[b,h] = sum_s w[s] * enc[b,s,h]   (enc[b,s,h] = encoder_out[s,b,h])
    for (int h = tid; h < H; h += 256) {
        float c = 0.0f;
        for (int s2 = 0; s2 < S; ++s2)
            c = fmaf(w_s[s2], enc_out[((size_t)s2 * B + b) * H + h], c);
        ctx[(size_t)b * H + h] = c * inv;
    }
}

// x16[b, :] = f16( [ wdv[b, inp[b], :], ctx[b,:], feed[b,:] ] )
__global__ void k_build_x(_Float16* __restrict__ x16,
                          const float* __restrict__ wdv,
                          const int* __restrict__ inp,
                          const float* __restrict__ ctx,
                          const float* __restrict__ feed) {
    long i = (long)blockIdx.x * blockDim.x + threadIdx.x;
    if (i >= (long)B * H3) return;
    int b = (int)(i / H3), c = (int)(i % H3);
    float val;
    if (c < H)            val = wdv[((size_t)b * V + inp[b]) * H + c];
    else if (c < 2 * H)   val = ctx[(size_t)b * H + (c - H)];
    else                  val = feed[(size_t)b * H + (c - 2 * H)];
    x16[i] = (_Float16)val;
}

// GRU gate math; updates h (f32) and h16, builds cat16 = f16([h_new, ctx]).
__global__ void k_gru_update(const float* __restrict__ gi, const float* __restrict__ gh,
                             float* __restrict__ h, _Float16* __restrict__ h16,
                             const float* __restrict__ ctx, _Float16* __restrict__ cat16) {
    int i = blockIdx.x * blockDim.x + threadIdx.x;
    if (i >= B * H) return;
    int b = i / H, hh = i % H;
    const size_t o = (size_t)b * H3;
    float r = sigmoidf_(gi[o + hh]           + gh[o + hh]);
    float z = sigmoidf_(gi[o + H + hh]       + gh[o + H + hh]);
    float n = tanhf(   gi[o + 2 * H + hh] + r * gh[o + 2 * H + hh]);
    float hn = (1.0f - z) * n + z * h[i];
    h[i] = hn;
    h16[i] = (_Float16)hn;
    cat16[(size_t)b * H2 + hh]     = (_Float16)hn;
    cat16[(size_t)b * H2 + H + hh] = (_Float16)ctx[i];
}

// One block per b (64 threads, v = tid < 37):
// sc[v] = tanh(wdv_proj[b,v,:] + hps[b,:]) . score_v ; mask ; write ; argmax -> inp
__launch_bounds__(64)
__global__ void k_score(const float* __restrict__ wdvproj,
                        const float* __restrict__ hps,
                        const float* __restrict__ score_v,
                        const unsigned char* __restrict__ mask,
                        float* __restrict__ out, int t, int* __restrict__ inp) {
    __shared__ float hp_s[H];
    __shared__ float sv_s[H];
    __shared__ float sc_s[64];
    const int b = blockIdx.x;
    const int tid = threadIdx.x;
    for (int i = tid; i < H; i += 64) { hp_s[i] = hps[(size_t)b * H + i]; sv_s[i] = score_v[i]; }
    __syncthreads();
    float sc = NEGV;
    if (tid < V) {
        if (mask[b * V + tid]) {
            sc = NEGV;
        } else {
            const float4* row4 = (const float4*)(wdvproj + ((size_t)(b * V + tid)) * H);
            float acc = 0.0f;
            for (int c = 0; c < H / 4; ++c) {
                float4 rv = row4[c];
                acc = fmaf(tanhf(rv.x + hp_s[c * 4 + 0]), sv_s[c * 4 + 0], acc);
                acc = fmaf(tanhf(rv.y + hp_s[c * 4 + 1]), sv_s[c * 4 + 1], acc);
                acc = fmaf(tanhf(rv.z + hp_s[c * 4 + 2]), sv_s[c * 4 + 2], acc);
                acc = fmaf(tanhf(rv.w + hp_s[c * 4 + 3]), sv_s[c * 4 + 3], acc);
            }
            sc = acc;
        }
        out[((size_t)t * B + b) * V + tid] = sc;
    }
    sc_s[tid] = sc;
    __syncthreads();
    if (tid == 0) {
        int best = 0; float bv = sc_s[0];
        for (int v = 1; v < V; ++v) if (sc_s[v] > bv) { bv = sc_s[v]; best = v; }  // first max wins
        inp[b] = best;
    }
}

// ---------------------------------------------------------------------------
// Host launcher
// ---------------------------------------------------------------------------
static inline size_t alignup(size_t x) { return (x + 255) & ~(size_t)255; }

extern "C" void kernel_launch(void* const* d_in, const int* in_sizes, int n_in,
                              void* d_out, int out_size, void* d_ws, size_t ws_size,
                              hipStream_t stream) {
    const float* enc_out  = (const float*)d_in[0];   // (S,B,H)
    const float* problem  = (const float*)d_in[1];   // (B,H)
    const int*   in_len   = (const int*)d_in[2];     // (B,)
    const int*   num_pos  = (const int*)d_in[3];     // (B,NUM)
    const int*   num_cnt  = (const int*)d_in[4];     // (B,)
    const float* embw     = (const float*)d_in[5];   // (22,H)
    const float* score_w  = (const float*)d_in[6];   // (2H,H)
    const float* score_b  = (const float*)d_in[7];   // (H,)
    const float* score_v  = (const float*)d_in[8];   // (H,)
    const float* attn_w   = (const float*)d_in[9];   // (2H,H)
    const float* attn_b   = (const float*)d_in[10];  // (H,)
    const float* attn_v   = (const float*)d_in[11];  // (H,)
    const float* gru_wih  = (const float*)d_in[12];  // (3H,3H)
    const float* gru_whh  = (const float*)d_in[13];  // (H,3H)
    const float* gru_bih  = (const float*)d_in[14];  // (3H,)
    const float* gru_bhh  = (const float*)d_in[15];  // (3H,)
    const float* feed_w   = (const float*)d_in[16];  // (2H,H)
    const float* feed_b   = (const float*)d_in[17];  // (H,)
    float* out = (float*)d_out;                      // (T1,B,V)

    // --- workspace carve (deterministic bump allocator) ---
    char* base = (char*)d_ws;
    size_t off = 0;
    auto carve = [&](size_t bytes) -> void* { void* p = base + off; off = alignup(off + bytes); return p; };

    _Float16* enc16     = (_Float16*)carve((size_t)B * S * H * 2);
    _Float16* encproj16 = (_Float16*)carve((size_t)B * S * H * 2);
    float*    wdv       = (float*)   carve((size_t)B * V * H * 4);
    _Float16* wdv16     = (_Float16*)carve((size_t)B * V * H * 2);
    float*    wdvproj   = (float*)   carve((size_t)B * V * H * 4);
    unsigned char* mask = (unsigned char*)carve((size_t)B * V);
    _Float16* awh_t     = (_Float16*)carve((size_t)H * H * 2);
    _Float16* awe_t     = (_Float16*)carve((size_t)H * H * 2);
    _Float16* swh_t     = (_Float16*)carve((size_t)H * H * 2);
    _Float16* swe_t     = (_Float16*)carve((size_t)H * H * 2);
    _Float16* wih_t     = (_Float16*)carve((size_t)H3 * H3 * 2);
    _Float16* whh_t     = (_Float16*)carve((size_t)H3 * H * 2);   // [N=3H][K=H]
    _Float16* feedw_t   = (_Float16*)carve((size_t)H * H2 * 2);   // [N=H][K=2H]
    float*    hbuf      = (float*)   carve((size_t)B * H * 4);
    _Float16* h16       = (_Float16*)carve((size_t)B * H * 2);
    float*    feed      = (float*)   carve((size_t)B * H * 4);
    float*    ctx       = (float*)   carve((size_t)B * H * 4);
    float*    hp        = (float*)   carve((size_t)B * H * 4);    // reused: h@aw_h, then h_new@sw_h
    int*      inp       = (int*)     carve((size_t)B * 4);
    _Float16* x16       = (_Float16*)carve((size_t)B * H3 * 2);
    float*    gi        = (float*)   carve((size_t)B * H3 * 4);
    float*    gh        = (float*)   carve((size_t)B * H3 * 4);
    _Float16* cat16     = (_Float16*)carve((size_t)B * H2 * 2);
    (void)ws_size; (void)n_in; (void)in_sizes; (void)out_size;

    auto g1 = [](long n) { return dim3((unsigned)((n + 255) / 256)); };

    // --- precompute: weight transpose+convert to f16 [N][K] ---
    k_wt_convert<<<g1((long)H * H),  256, 0, stream>>>(awh_t,  attn_w,          H,  H);   // aw_h = attn_w[:H]
    k_wt_convert<<<g1((long)H * H),  256, 0, stream>>>(awe_t,  attn_w + H * H,  H,  H);   // aw_e = attn_w[H:]
    k_wt_convert<<<g1((long)H * H),  256, 0, stream>>>(swh_t,  score_w,         H,  H);
    k_wt_convert<<<g1((long)H * H),  256, 0, stream>>>(swe_t,  score_w + H * H, H,  H);
    k_wt_convert<<<g1((long)H3 * H3),256, 0, stream>>>(wih_t,  gru_wih,         H3, H3);
    k_wt_convert<<<g1((long)H * H3), 256, 0, stream>>>(whh_t,  gru_whh,         H,  H3);
    k_wt_convert<<<g1((long)H2 * H), 256, 0, stream>>>(feedw_t,feed_w,          H2, H);

    k_convert_enc<<<g1((long)B * S * H), 256, 0, stream>>>(enc16, enc_out);
    k_build_wdv  <<<g1((long)B * V * H), 256, 0, stream>>>(wdv, wdv16, embw, enc_out, num_pos, num_cnt);
    k_build_mask <<<g1((long)B * V),     256, 0, stream>>>(mask, num_cnt);
    k_init       <<<g1((long)B * H),     256, 0, stream>>>(hbuf, h16, feed, inp, problem);

    // enc_proj16 = enc16 @ aw_e + attn_b   (M=32768, K=768, N=768) -> f16
    gemm_wmma_f16<<<dim3(H / 64, (B * S) / 32), 128, 0, stream>>>(
        enc16, awe_t, attn_b, nullptr, encproj16, B * S, H, H, 0);
    // wdv_proj = wdv16 @ sw_e + score_b    (M=4736, K=768, N=768) -> f32
    gemm_wmma_f16<<<dim3(H / 64, (B * V) / 32), 128, 0, stream>>>(
        wdv16, swe_t, score_b, wdvproj, nullptr, B * V, H, H, 0);

    // --- decode loop (sequential recurrence) ---
    for (int t = 0; t < T1; ++t) {
        // hp = h @ aw_h  (128x768 * 768x768)
        gemm_wmma_f16<<<dim3(H / 64, B / 32), 128, 0, stream>>>(
            h16, awh_t, nullptr, hp, nullptr, B, H, H, 0);
        // attention -> ctx
        k_attention<<<dim3(B), 256, 0, stream>>>(encproj16, hp, attn_v, in_len, enc_out, ctx);
        // x16 = [emb, ctx, feed]
        k_build_x<<<g1((long)B * H3), 256, 0, stream>>>(x16, wdv, inp, ctx, feed);
        // gi = x @ gru_wih + bih  (128x2304 * 2304x2304)
        gemm_wmma_f16<<<dim3(H3 / 64, B / 32), 128, 0, stream>>>(
            x16, wih_t, gru_bih, gi, nullptr, B, H3, H3, 0);
        // gh = h @ gru_whh + bhh  (128x768 * 768x2304)
        gemm_wmma_f16<<<dim3(H3 / 64, B / 32), 128, 0, stream>>>(
            h16, whh_t, gru_bhh, gh, nullptr, B, H3, H, 0);
        // GRU gates -> h, h16, cat16
        k_gru_update<<<g1((long)B * H), 256, 0, stream>>>(gi, gh, hbuf, h16, ctx, cat16);
        // feed = tanh([h_new, ctx] @ feed_w + feed_b)  (128x1536 * 1536x768)
        gemm_wmma_f16<<<dim3(H / 64, B / 32), 128, 0, stream>>>(
            cat16, feedw_t, feed_b, feed, nullptr, B, H, H2, 1);
        // hps = h_new @ sw_h  (reuse hp buffer)
        gemm_wmma_f16<<<dim3(H / 64, B / 32), 128, 0, stream>>>(
            h16, swh_t, nullptr, hp, nullptr, B, H, H, 0);
        // scores + argmax -> inp
        k_score<<<dim3(B), 64, 0, stream>>>(wdvproj, hp, score_v, mask, out, t, inp);
    }
}
